// GraphConvolutionLayer_GCN_23605140259235
// MI455X (gfx1250) — matrix operand order repaired and checked by reading, not compiled, not run on hardware
//
#include <hip/hip_runtime.h>
#include <stdint.h>
#include <stddef.h>

#define N_NODES 16384
#define D 512

typedef float  f4    __attribute__((ext_vector_type(4)));
typedef float  v8f   __attribute__((ext_vector_type(8)));
typedef __bf16 v16bf __attribute__((ext_vector_type(16)));
typedef __bf16 bf8   __attribute__((ext_vector_type(8)));
typedef unsigned int u32x4 __attribute__((ext_vector_type(4)));
typedef unsigned int u32x2 __attribute__((ext_vector_type(2)));

// f32 -> bf16 round-to-nearest-even
__device__ __forceinline__ unsigned short f2bf(float f) {
    unsigned u = __builtin_bit_cast(unsigned, f);
    u += 0x7FFFu + ((u >> 16) & 1u);
    return (unsigned short)(u >> 16);
}

__device__ __forceinline__ float bcastf(float v, int l) {
    return __builtin_bit_cast(float,
        __builtin_amdgcn_readlane(__builtin_bit_cast(int, v), l));
}

// ---------------- Kernel 0: W (f32) -> bf16 ----------------
__global__ __launch_bounds__(256) void convert_w_bf16(
    const float* __restrict__ W, unsigned short* __restrict__ Wb) {
    int idx = blockIdx.x * 256 + threadIdx.x;
    Wb[idx] = f2bf(W[idx]);
}

// ---------------- Kernel 1: sparse aggregate ----------------
// pooled[i,:] = (sum_j A[i,j]*x[j,:] + x[i,:]) / deg[i] + x[i,:]   -> bf16
// One wave per row. Lane l owns cols {c*128 + l*4 .. +3 : c=0..3} (16 f32 acc).
// Adjacency is streamed non-temporally (1 GiB, HBM-bound); x stays L2-resident.
__global__ __launch_bounds__(256) void gcn_aggregate(
    const float* __restrict__ x, const float* __restrict__ adj,
    const float* __restrict__ deg, unsigned short* __restrict__ poolb) {
    const int wave = threadIdx.x >> 5;
    const int lane = threadIdx.x & 31;
    const int row  = blockIdx.x * 8 + wave;
    const size_t rowoff = (size_t)row * N_NODES;

    f4 acc0 = {0.f,0.f,0.f,0.f};
    f4 acc1 = acc0, acc2 = acc0, acc3 = acc0;

    for (int t = 0; t < N_NODES / 128; ++t) {
        const f4* ap = (const f4*)(adj + rowoff + (size_t)t * 128) + lane;
        if (t + 4 < N_NODES / 128)
            __builtin_prefetch((const float*)ap + 512, 0, 0);   // global_prefetch_b8
        f4 a4 = __builtin_nontemporal_load(ap);                 // stream A past L2
        bool nz = (a4.x != 0.f) | (a4.y != 0.f) | (a4.z != 0.f) | (a4.w != 0.f);
        unsigned m = __builtin_amdgcn_ballot_w32(nz);
        while (m) {
            int b = __builtin_ctz(m);
            m &= m - 1;
            float wv0 = bcastf(a4.x, b), wv1 = bcastf(a4.y, b);
            float wv2 = bcastf(a4.z, b), wv3 = bcastf(a4.w, b);
            float wv[4] = {wv0, wv1, wv2, wv3};
            int jbase = t * 128 + b * 4;
            #pragma unroll
            for (int k = 0; k < 4; ++k) {
                if (wv[k] != 0.f) {                 // scalar (SGPR) branch
                    const f4* xr = (const f4*)(x + (size_t)(jbase + k) * D);
                    acc0 += wv[k] * xr[lane];
                    acc1 += wv[k] * xr[lane + 32];
                    acc2 += wv[k] * xr[lane + 64];
                    acc3 += wv[k] * xr[lane + 96];
                }
            }
        }
    }

    const float inv = 1.0f / deg[row];
    const f4* xi = (const f4*)(x + (size_t)row * D);
    f4 accs[4] = {acc0, acc1, acc2, acc3};
    #pragma unroll
    for (int c = 0; c < 4; ++c) {
        f4 xv = xi[c * 32 + lane];
        f4 p  = (accs[c] + xv) * inv + xv;
        unsigned lo = (unsigned)f2bf(p.x) | ((unsigned)f2bf(p.y) << 16);
        unsigned hi = (unsigned)f2bf(p.z) | ((unsigned)f2bf(p.w) << 16);
        u32x2 pk = {lo, hi};
        *(u32x2*)&poolb[(size_t)row * D + c * 128 + lane * 4] = pk;
    }
}

// ---------------- Kernel 2: out = relu(pooled @ W + b), bf16 WMMA ----------------
// Block tile 128(M) x 64(N), BK=32. 8 waves: wave w -> rows w*16..+15, 4 N-tiles.
// Double-buffered LDS (one barrier per K-step); next tile's global loads issued
// before the WMMA chain; all 4 B-fragments preloaded so WMMAs issue back-to-back.
__global__ __launch_bounds__(256) void gcn_linear(
    const unsigned short* __restrict__ poolb,
    const unsigned short* __restrict__ Wb,
    const float* __restrict__ bias, float* __restrict__ out) {
    __shared__ unsigned short As[2][128][40];  // pitch 20 words: 16B aligned frags
    __shared__ unsigned short Bs[2][64][40];   // W tile transposed: Bs[.][n][k]

    const int tid  = threadIdx.x;
    const int wave = tid >> 5;
    const int lane = tid & 31;
    const int g    = lane >> 4;       // K half-group
    const int ln   = lane & 15;
    const int bm   = blockIdx.x * 128;
    const int bn   = blockIdx.y * 64;

    // per-thread staging coordinates
    const int ar = tid >> 1, ah = tid & 1;              // A: 2 threads/row, 32B each
    const int wk = tid & 31, wn = (tid >> 5) * 8;       // B: thread -> (k, 8-col chunk)
    const unsigned short* pa = poolb + (size_t)(bm + ar) * D + ah * 16;
    const unsigned short* pw = Wb + (size_t)wk * D + bn + wn;

    v8f acc[4] = {};
    u32x4 qa0, qa1, qb;

    // prologue: fetch tile 0 and commit to buffer 0
    qa0 = ((const u32x4*)pa)[0];
    qa1 = ((const u32x4*)pa)[1];
    qb  = ((const u32x4*)pw)[0];
    *(u32x4*)&As[0][ar][ah * 16]     = qa0;
    *(u32x4*)&As[0][ar][ah * 16 + 8] = qa1;
    #pragma unroll
    for (int j = 0; j < 8; ++j) {
        unsigned w = qb[j >> 1];
        Bs[0][wn + j][wk] = (unsigned short)((j & 1) ? (w >> 16) : (w & 0xFFFFu));
    }
    __syncthreads();

    for (int kb = 0; kb < D / 32; ++kb) {
        const int cur = kb & 1;

        // issue next tile's global loads early (latency hidden under WMMA)
        if (kb + 1 < D / 32) {
            const unsigned short* pa2 = pa + (kb + 1) * 32;
            const unsigned short* pw2 = pw + (size_t)(kb + 1) * 32 * D;
            qa0 = ((const u32x4*)pa2)[0];
            qa1 = ((const u32x4*)pa2)[1];
            qb  = ((const u32x4*)pw2)[0];
        }

        // A fragment per ISA 7.12.2: lanes 0-15 -> K {0-7,16-23}, lanes 16-31 -> +8
        int m = wave * 16 + ln;
        bf8 alo = __builtin_bit_cast(bf8, *(const u32x4*)&As[cur][m][g * 8]);
        bf8 ahi = __builtin_bit_cast(bf8, *(const u32x4*)&As[cur][m][16 + g * 8]);
        v16bf a = __builtin_shufflevector(alo, ahi,
                    0,1,2,3,4,5,6,7,8,9,10,11,12,13,14,15);

        // preload all 4 B fragments, then run the WMMA chain back-to-back
        v16bf bfr[4];
        #pragma unroll
        for (int t = 0; t < 4; ++t) {
            int n = t * 16 + ln;
            bf8 blo = __builtin_bit_cast(bf8, *(const u32x4*)&Bs[cur][n][g * 8]);
            bf8 bhi = __builtin_bit_cast(bf8, *(const u32x4*)&Bs[cur][n][16 + g * 8]);
            bfr[t] = __builtin_shufflevector(blo, bhi,
                        0,1,2,3,4,5,6,7,8,9,10,11,12,13,14,15);
        }
        #pragma unroll
        for (int t = 0; t < 4; ++t) {
            acc[t] = __builtin_amdgcn_wmma_f32_16x16x32_bf16(
                false, a, false, bfr[t], (short)0, acc[t], false, false);
        }

        // commit next tile to the alternate buffer; single barrier per step
        if (kb + 1 < D / 32) {
            const int nxt = 1 - cur;
            *(u32x4*)&As[nxt][ar][ah * 16]     = qa0;
            *(u32x4*)&As[nxt][ar][ah * 16 + 8] = qa1;
            #pragma unroll
            for (int j = 0; j < 8; ++j) {
                unsigned w = qb[j >> 1];
                Bs[nxt][wn + j][wk] =
                    (unsigned short)((j & 1) ? (w >> 16) : (w & 0xFFFFu));
            }
            __syncthreads();
        }
    }

    // Epilogue: bias + ReLU, f32 out. D layout: M = g*8 + v, N = ln.
    #pragma unroll
    for (int t = 0; t < 4; ++t) {
        int col = bn + t * 16 + ln;
        float bv = bias[col];
        #pragma unroll
        for (int v = 0; v < 8; ++v) {
            int row = bm + wave * 16 + g * 8 + v;
            float val = acc[t][v] + bv;
            out[(size_t)row * D + col] = val > 0.f ? val : 0.f;
        }
    }
}

extern "C" void kernel_launch(void* const* d_in, const int* in_sizes, int n_in,
                              void* d_out, int out_size, void* d_ws, size_t ws_size,
                              hipStream_t stream) {
    (void)in_sizes; (void)n_in; (void)out_size; (void)ws_size;
    const float* x   = (const float*)d_in[0];
    const float* adj = (const float*)d_in[1];
    const float* deg = (const float*)d_in[2];
    const float* W   = (const float*)d_in[3];
    const float* b   = (const float*)d_in[4];
    float* out = (float*)d_out;

    unsigned short* Wb    = (unsigned short*)d_ws;                            // 512 KB
    unsigned short* poolb = (unsigned short*)((char*)d_ws + (size_t)D * D * 2); // 16 MB

    convert_w_bf16<<<D * D / 256, 256, 0, stream>>>(W, Wb);
    gcn_aggregate<<<N_NODES / 8, 256, 0, stream>>>(x, adj, deg, poolb);
    gcn_linear<<<dim3(N_NODES / 128, D / 64), 256, 0, stream>>>(poolb, Wb, b, out);
}
